// RetNet_79319456022754
// MI455X (gfx1250) — compile-verified, matrix operand order checked by reading, not run on hardware
//
#include <hip/hip_runtime.h>
#include <hip/hip_bf16.h>

// ---------------- types ----------------
typedef __bf16 bf16;
typedef bf16  v16bf __attribute__((ext_vector_type(16)));
typedef float v8f   __attribute__((ext_vector_type(8)));
typedef int   v4i_vec __attribute__((vector_size(16)));  // for async-LDS builtin

union FragAB { v16bf v; bf16 e[16]; uint4 q[2]; };
union FragC  { v8f v; float e[8]; };

__device__ __forceinline__ v8f wmma_bf(v16bf a, v16bf b, v8f c) {
  // D = A(16x32 bf16) * B(32x16 bf16) + C(16x16 f32)
  return __builtin_amdgcn_wmma_f32_16x16x32_bf16(false, a, false, b, (short)0, c,
                                                 false, false);
}

// CDNA5 async global->LDS staging (guarded: falls back to load+ds_store)
#if defined(__has_builtin)
#if __has_builtin(__builtin_amdgcn_global_load_async_to_lds_b128) && \
    __has_builtin(__builtin_amdgcn_s_wait_asynccnt)
#define USE_ASYNC_LDS 1
#endif
#endif
#ifndef USE_ASYNC_LDS
#define USE_ASYNC_LDS 0
#endif

// model constants
constexpr int kD  = 1024;
constexpr int kH  = 16;
constexpr int kDK = 64;
constexpr int kDV = 128;
constexpr int kVD = kH * kDV;  // 2048
constexpr int kF  = 4096;
constexpr int kT  = 8 * 2048;  // B*S = 16384 (reference flattens batch)
constexpr int kC  = 64;
constexpr float kEPS = 1e-5f;

// ---------------- f32 -> bf16 transposed convert: d[n*K+k] = s[k*N+n] ----------------
__global__ void cvt_t_bf16_kernel(const float* __restrict__ s, bf16* __restrict__ d,
                                  int K, int N) {
  int idx = blockIdx.x * blockDim.x + threadIdx.x;
  if (idx >= K * N) return;
  int k = idx / N, n = idx % N;
  d[(size_t)n * K + k] = (bf16)s[idx];
}

// ---------------- layernorm (row of 1024) -> f32 and/or bf16 ----------------
__global__ __launch_bounds__(256)
void layernorm_kernel(const float* __restrict__ x, const float* __restrict__ w,
                      const float* __restrict__ b, float* __restrict__ outf,
                      bf16* __restrict__ outb) {
  __shared__ float red[256];
  const int row = blockIdx.x, tid = threadIdx.x;
  const float* xr = x + (size_t)row * kD;
  float4 xv = *(const float4*)(xr + tid * 4);

  red[tid] = xv.x + xv.y + xv.z + xv.w;
  __syncthreads();
  for (int off = 128; off; off >>= 1) {
    if (tid < off) red[tid] += red[tid + off];
    __syncthreads();
  }
  const float mu = red[0] * (1.0f / kD);
  __syncthreads();

  float dx = xv.x - mu, dy = xv.y - mu, dz = xv.z - mu, dw = xv.w - mu;
  red[tid] = dx * dx + dy * dy + dz * dz + dw * dw;
  __syncthreads();
  for (int off = 128; off; off >>= 1) {
    if (tid < off) red[tid] += red[tid + off];
    __syncthreads();
  }
  const float rs = rsqrtf(red[0] * (1.0f / kD) + kEPS);

  float xe[4] = {xv.x, xv.y, xv.z, xv.w};
#pragma unroll
  for (int j = 0; j < 4; j++) {
    int c = tid * 4 + j;
    float y = (xe[j] - mu) * rs * w[c] + b[c];
    if (outf) outf[(size_t)row * kD + c] = y;
    if (outb) outb[(size_t)row * kD + c] = (bf16)y;
  }
}

// ---------------- rotary embedding, f32 in -> bf16 out, fused scale ----------------
__global__ void rotary_kernel(const float* __restrict__ src, bf16* __restrict__ dst,
                              float scale) {
  int idx = blockIdx.x * blockDim.x + threadIdx.x;  // over T*H*32
  int t = idx >> 9;
  int rem = idx & 511;
  int h = rem >> 5;
  int i = rem & 31;
  size_t base = ((size_t)t * kH + h) * kDK;
  float x1 = src[base + i];
  float x2 = src[base + 32 + i];
  float inv = powf(10000.0f, -(float)i / 32.0f);
  float ang = (float)t * inv;
  float c = cosf(ang), s = sinf(ang);
  dst[base + i]      = (bf16)((x1 * c - x2 * s) * scale);
  dst[base + 32 + i] = (bf16)((x2 * c + x1 * s) * scale);
}

// ---------------- bf16 WMMA GEMM: C[M,N] = A[M,K] * Bt[N,K]^T ----------------
// A: row-major activations (LDS double-buffered, async staged).
// Bt: TRANSPOSED weights [N][K] -> B fragments load straight from global (L2).
// epi: 0 f32 store | 1 bf16 store | 2 bias+GELU->bf16 | 3 bias+residual->f32
#define BM 64
#define BN 128
#define BK 32
__global__ __launch_bounds__(256)
void gemm_bf16_kernel(const bf16* __restrict__ A, const bf16* __restrict__ Bt,
                      float* __restrict__ Cf, bf16* __restrict__ Cb,
                      const float* __restrict__ bias, const float* __restrict__ resid,
                      int M, int N, int K, int epi) {
  __shared__ bf16 As[2][BM * BK];
  const int tid = threadIdx.x;
  const int wave = tid >> 5, lane = tid & 31;
  const int r = lane & 15, kh = lane >> 4;
  const int m0 = blockIdx.y * BM, n0 = blockIdx.x * BN;
  const int arow = tid >> 2, acol = (tid & 3) * 8;  // A stage: 8 bf16/thread

  const int coln = n0 + wave * 16 + r;              // this wave's B column
  const bf16* Bcol = Bt + (size_t)coln * K;

  FragC acc[4];
#pragma unroll
  for (int i = 0; i < 4; i++) acc[i].v = (v8f){0, 0, 0, 0, 0, 0, 0, 0};

  auto stage = [&](int b, int k0) {
    const bf16* src = A + (size_t)(m0 + arow) * K + k0 + acol;
    bf16* dst = &As[b][arow * BK + acol];
#if USE_ASYNC_LDS
    __builtin_amdgcn_global_load_async_to_lds_b128(
        (__attribute__((address_space(1))) v4i_vec*)src,
        (__attribute__((address_space(3))) v4i_vec*)dst, 0, 0);
#else
    uint4 av = *(const uint4*)src;
    *(uint4*)dst = av;
#endif
  };
  auto stage_wait = [&]() {
#if USE_ASYNC_LDS
    __builtin_amdgcn_s_wait_asynccnt(0);
#endif
  };

  stage(0, 0);
  stage_wait();
  __syncthreads();

  int buf = 0;
  for (int k0 = 0; k0 < K; k0 += BK, buf ^= 1) {
    if (k0 + BK < K) stage(buf ^ 1, k0 + BK);
    if (k0 + 2 * BK < K)
      __builtin_prefetch(A + (size_t)(m0 + arow) * K + k0 + 2 * BK + acol, 0, 1);

    // B fragment: contiguous-in-K from transposed weights (global/L2)
    FragAB bfr;
    bfr.q[0] = *(const uint4*)(Bcol + k0 + kh * 8);
    bfr.q[1] = *(const uint4*)(Bcol + k0 + 16 + kh * 8);

#pragma unroll
    for (int mt = 0; mt < 4; mt++) {
      FragAB af;
      const int rowi = mt * 16 + r;
      af.q[0] = *(const uint4*)(&As[buf][rowi * BK + kh * 8]);
      af.q[1] = *(const uint4*)(&As[buf][rowi * BK + 16 + kh * 8]);
      acc[mt].v = wmma_bf(af.v, bfr.v, acc[mt].v);
    }
    stage_wait();
    __syncthreads();
  }

#pragma unroll
  for (int mt = 0; mt < 4; mt++) {
#pragma unroll
    for (int e = 0; e < 8; e++) {
      int rowm = m0 + mt * 16 + e + ((lane >= 16) ? 8 : 0);
      float vv = acc[mt].e[e];
      if (epi >= 2) vv += bias[coln];
      if (epi == 2) vv = 0.5f * vv * (1.0f + erff(vv * 0.70710678118654752f));
      size_t oi = (size_t)rowm * N + coln;
      if (epi == 3) vv += resid[oi];
      if (epi == 0 || epi == 3) Cf[oi] = vv;
      else Cb[oi] = (bf16)vv;
    }
  }
}

// ---------------- retention chunk scan: one workgroup per head ----------------
__global__ __launch_bounds__(256)
void retention_kernel(const bf16* __restrict__ q, const bf16* __restrict__ k,
                      const bf16* __restrict__ v, float* __restrict__ o) {
  __shared__ float St[kDV * kDK];   // f32 state, TRANSPOSED: St[col][kidx], 32 KB
  __shared__ bf16 vT[kDV * kC];     // v chunk transposed [col][j], 16 KB
  __shared__ bf16 scr[kC * kC];     // Albs (phases A/B) then k^T*kdec (phase C), 8 KB
  __shared__ float qdec[kC], kdec[kC];

  const int h = blockIdx.x;
  const int tid = threadIdx.x, wave = tid >> 5, lane = tid & 31;
  const int r = lane & 15, kh = lane >> 4, hi8 = (lane >= 16) ? 8 : 0;
  const int col = wave * 16 + (lane & 15);  // this wave's value-dim column
  const float gamma = 1.0f - exp2f(-5.0f - (float)h);
  const float lg = log2f(gamma);
  const float cdec = exp2f((float)kC * lg);
  if (tid < kC) {
    qdec[tid] = exp2f((float)(tid + 1) * lg);
    kdec[tid] = exp2f((float)(kC - 1 - tid) * lg);
  }
  for (int i = tid; i < kDV * kDK; i += 256) St[i] = 0.0f;
  __syncthreads();

  const size_t qs = (size_t)kH * kDK;
  const size_t vs = (size_t)kH * kDV;

  for (int nc = 0; nc < kT / kC; nc++) {
    const int t0 = nc * kC;

    // ---- stage v chunk transposed: vT[col][j] ----
#pragma unroll
    for (int rep = 0; rep < 2; rep++) {
      int cid = tid + rep * 256;           // 0..511
      int j = cid >> 3, coff = (cid & 7) * 16;
      FragAB tmp;
      const bf16* vp = v + (size_t)(t0 + j) * vs + h * kDV + coff;
      tmp.q[0] = *(const uint4*)(vp);
      tmp.q[1] = *(const uint4*)(vp + 8);
#pragma unroll
      for (int e = 0; e < 16; e++) vT[(coff + e) * kC + j] = tmp.e[e];
    }

    // ---- Phase A: A = dmask * (q k^T) -> scr (bf16) ----
#pragma unroll
    for (int ti = 0; ti < 2; ti++) {
      int tile = wave * 2 + ti;
      int it = tile >> 2, jt = tile & 3;
      FragC d; d.v = (v8f){0, 0, 0, 0, 0, 0, 0, 0};
#pragma unroll
      for (int ks = 0; ks < kDK; ks += 32) {
        FragAB af, bfr;
        const bf16* qp = q + (size_t)(t0 + it * 16 + r) * qs + h * kDK + ks;
        af.q[0] = *(const uint4*)(qp + kh * 8);
        af.q[1] = *(const uint4*)(qp + 16 + kh * 8);
        const bf16* kp = k + (size_t)(t0 + jt * 16 + r) * qs + h * kDK + ks;
        bfr.q[0] = *(const uint4*)(kp + kh * 8);
        bfr.q[1] = *(const uint4*)(kp + 16 + kh * 8);
        d.v = wmma_bf(af.v, bfr.v, d.v);
      }
#pragma unroll
      for (int e = 0; e < 8; e++) {
        int ii = it * 16 + e + hi8;
        int jj = jt * 16 + (lane & 15);
        float val = (ii >= jj) ? d.e[e] * exp2f((float)(ii - jj) * lg) : 0.0f;
        scr[ii * kC + jj] = (bf16)val;
      }
    }
    __syncthreads();

    // ---- Phase B: o = qdec*(q @ S) + A @ v ----
#pragma unroll
    for (int mt = 0; mt < 4; mt++) {
      const int rowi = mt * 16 + r;
      FragC accI; accI.v = (v8f){0, 0, 0, 0, 0, 0, 0, 0};
      // inter-chunk: raw q @ S  (per-row qdec applied after)
#pragma unroll
      for (int ks = 0; ks < kDK; ks += 32) {
        FragAB af, bfr;
        const bf16* qp = q + (size_t)(t0 + rowi) * qs + h * kDK + ks;
        af.q[0] = *(const uint4*)(qp + kh * 8);
        af.q[1] = *(const uint4*)(qp + 16 + kh * 8);
        const float* sp = &St[col * kDK + ks + kh * 8];
        float4 s0 = *(const float4*)(sp);
        float4 s1 = *(const float4*)(sp + 4);
        float4 s2 = *(const float4*)(sp + 16);
        float4 s3 = *(const float4*)(sp + 20);
        bfr.e[0] = (bf16)s0.x;  bfr.e[1] = (bf16)s0.y;
        bfr.e[2] = (bf16)s0.z;  bfr.e[3] = (bf16)s0.w;
        bfr.e[4] = (bf16)s1.x;  bfr.e[5] = (bf16)s1.y;
        bfr.e[6] = (bf16)s1.z;  bfr.e[7] = (bf16)s1.w;
        bfr.e[8] = (bf16)s2.x;  bfr.e[9] = (bf16)s2.y;
        bfr.e[10] = (bf16)s2.z; bfr.e[11] = (bf16)s2.w;
        bfr.e[12] = (bf16)s3.x; bfr.e[13] = (bf16)s3.y;
        bfr.e[14] = (bf16)s3.z; bfr.e[15] = (bf16)s3.w;
        accI.v = wmma_bf(af.v, bfr.v, accI.v);
      }
      FragC accO;
#pragma unroll
      for (int e = 0; e < 8; e++)
        accO.e[e] = accI.e[e] * qdec[mt * 16 + e + hi8];
      // intra-chunk: A @ v
#pragma unroll
      for (int js = 0; js < kC; js += 32) {
        FragAB af, bfr;
        af.q[0] = *(const uint4*)(&scr[rowi * kC + js + kh * 8]);
        af.q[1] = *(const uint4*)(&scr[rowi * kC + js + 16 + kh * 8]);
        bfr.q[0] = *(const uint4*)(&vT[col * kC + js + kh * 8]);
        bfr.q[1] = *(const uint4*)(&vT[col * kC + js + 16 + kh * 8]);
        accO.v = wmma_bf(af.v, bfr.v, accO.v);
      }
#pragma unroll
      for (int e = 0; e < 8; e++) {
        int rm = mt * 16 + e + hi8;
        o[(size_t)(t0 + rm) * vs + h * kDV + col] = accO.e[e];
      }
    }
    __syncthreads();

    // ---- stage decayed k^T into scr: scr[kidx][c] = k[t0+c][kidx]*kdec[c] ----
    {
      int c = tid >> 2, koff = (tid & 3) * 16;
      FragAB tk;
      const bf16* kp = k + (size_t)(t0 + c) * qs + h * kDK + koff;
      tk.q[0] = *(const uint4*)(kp);
      tk.q[1] = *(const uint4*)(kp + 8);
      float kd = kdec[c];
#pragma unroll
      for (int e = 0; e < 16; e++)
        scr[(koff + e) * kC + c] = (bf16)((float)tk.e[e] * kd);
    }
    __syncthreads();

    // ---- Phase C: S = cdec*S + (k*kdec)^T @ v ----
#pragma unroll
    for (int kt = 0; kt < 4; kt++) {
      const int rowk = kt * 16 + r;
      FragC accS; accS.v = (v8f){0, 0, 0, 0, 0, 0, 0, 0};
#pragma unroll
      for (int cs = 0; cs < kC; cs += 32) {
        FragAB af, bfr;
        af.q[0] = *(const uint4*)(&scr[rowk * kC + cs + kh * 8]);
        af.q[1] = *(const uint4*)(&scr[rowk * kC + cs + 16 + kh * 8]);
        bfr.q[0] = *(const uint4*)(&vT[col * kC + cs + kh * 8]);
        bfr.q[1] = *(const uint4*)(&vT[col * kC + cs + 16 + kh * 8]);
        accS.v = wmma_bf(af.v, bfr.v, accS.v);
      }
#pragma unroll
      for (int e = 0; e < 8; e++) {
        int rk = kt * 16 + e + hi8;
        float* sp = &St[col * kDK + rk];
        *sp = cdec * *sp + accS.e[e];
      }
    }
    __syncthreads();
  }
}

// ---------------- groupnorm (RMS over DV) + SiLU gate -> bf16 ----------------
__global__ __launch_bounds__(256)
void gn_gate_kernel(const float* __restrict__ o, const bf16* __restrict__ g,
                    const float* __restrict__ gnw, bf16* __restrict__ out) {
  const int tid = threadIdx.x, wave = tid >> 5, lane = tid & 31;
  const int wid = blockIdx.x * 8 + wave;  // one wave per (t,h)
  const int t = wid >> 4, h = wid & 15;
  const size_t base = ((size_t)t * kH + h) * kDV;
  float ov[4];
  float s = 0.0f;
#pragma unroll
  for (int j = 0; j < 4; j++) {
    int dv = lane + 32 * j;
    ov[j] = o[base + dv];
    s += ov[j] * ov[j];
  }
#pragma unroll
  for (int off = 16; off; off >>= 1) s += __shfl_xor(s, off);
  const float rs = rsqrtf(s * (1.0f / kDV) + kEPS);
#pragma unroll
  for (int j = 0; j < 4; j++) {
    int dv = lane + 32 * j;
    float gv = (float)g[base + dv];
    float sig = 1.0f / (1.0f + expf(-gv));
    out[base + dv] = (bf16)(ov[j] * rs * gnw[dv] * gv * sig);
  }
}

// ---------------- host-side launch ----------------
extern "C" void kernel_launch(void* const* d_in, const int* in_sizes, int n_in,
                              void* d_out, int out_size, void* d_ws, size_t ws_size,
                              hipStream_t stream) {
  const float* X      = (const float*)d_in[0];
  const float* ln1_w  = (const float*)d_in[1];
  const float* ln1_b  = (const float*)d_in[2];
  const float* ln2_w  = (const float*)d_in[3];
  const float* ln2_b  = (const float*)d_in[4];
  const float* wq     = (const float*)d_in[5];
  const float* wk     = (const float*)d_in[6];
  const float* wv     = (const float*)d_in[7];
  const float* wg     = (const float*)d_in[8];
  const float* gn_w   = (const float*)d_in[9];
  const float* wo     = (const float*)d_in[10];
  const float* ffn_w1 = (const float*)d_in[11];
  const float* ffn_b1 = (const float*)d_in[12];
  const float* ffn_w2 = (const float*)d_in[13];
  const float* ffn_b2 = (const float*)d_in[14];

  char* ws = (char*)d_ws;
  size_t off = 0;
  auto alloc = [&](size_t bytes) -> void* {
    void* p = ws + off;
    off = (off + bytes + 255) & ~(size_t)255;
    return p;
  };

  bf16*  h_bf   = (bf16*)alloc((size_t)kT * kD * 2);
  float* q_f    = (float*)alloc((size_t)kT * (kH * kDK) * 4);
  float* k_f    = (float*)alloc((size_t)kT * (kH * kDK) * 4);
  bf16*  qr_bf  = (bf16*)alloc((size_t)kT * (kH * kDK) * 2);
  bf16*  kr_bf  = (bf16*)alloc((size_t)kT * (kH * kDK) * 2);
  bf16*  v_bf   = (bf16*)alloc((size_t)kT * kVD * 2);
  bf16*  g_bf   = (bf16*)alloc((size_t)kT * kVD * 2);
  float* o_f    = (float*)alloc((size_t)kT * kVD * 4);
  bf16*  gated  = (bf16*)alloc((size_t)kT * kVD * 2);
  float* x_buf  = (float*)alloc((size_t)kT * kD * 4);
  bf16*  wq_bf  = (bf16*)alloc((size_t)kD * 1024 * 2);   // all weight bufs transposed [N][K]
  bf16*  wk_bf  = (bf16*)alloc((size_t)kD * 1024 * 2);
  bf16*  wv_bf  = (bf16*)alloc((size_t)kD * kVD * 2);
  bf16*  wg_bf  = (bf16*)alloc((size_t)kD * kVD * 2);
  bf16*  wo_bf  = (bf16*)alloc((size_t)kVD * kD * 2);
  bf16*  w1_bf  = (bf16*)alloc((size_t)kD * kF * 2);
  bf16*  w2_bf  = (bf16*)alloc((size_t)kF * kD * 2);

  // aliases over dead buffers
  float* y_f     = q_f;          // wo output (q dead after rotary)
  float* h2_f    = k_f;          // ln2 f32 output (k dead after rotary)
  bf16*  h2_bf   = h_bf;         // ln2 bf16 output (h dead after projections)
  bf16*  ffn1_bf = (bf16*)o_f;   // ffn1 output (o dead after gating)

  auto cvt_t = [&](const float* s, bf16* d, int K, int N) {
    cvt_t_bf16_kernel<<<(K * N + 255) / 256, 256, 0, stream>>>(s, d, K, N);
  };

  for (int i = 0; i < 2; i++) {
    const float* xin = (i == 0) ? X : x_buf;
    float* xout = (i == 1) ? (float*)d_out : x_buf;

    // weights -> transposed bf16 [N][K]
    cvt_t(wq + (size_t)i * kD * 1024, wq_bf, kD, 1024);
    cvt_t(wk + (size_t)i * kD * 1024, wk_bf, kD, 1024);
    cvt_t(wv + (size_t)i * kD * kVD, wv_bf, kD, kVD);
    cvt_t(wg + (size_t)i * kD * kVD, wg_bf, kD, kVD);
    cvt_t(wo + (size_t)i * kVD * kD, wo_bf, kVD, kD);
    cvt_t(ffn_w1 + (size_t)i * kD * kF, w1_bf, kD, kF);
    cvt_t(ffn_w2 + (size_t)i * kF * kD, w2_bf, kF, kD);

    // LN1 -> h (bf16)
    layernorm_kernel<<<kT, 256, 0, stream>>>(xin, ln1_w + i * kD, ln1_b + i * kD,
                                             nullptr, h_bf);
    // projections
    gemm_bf16_kernel<<<dim3(1024 / BN, kT / BM), 256, 0, stream>>>(
        h_bf, wq_bf, q_f, nullptr, nullptr, nullptr, kT, 1024, kD, 0);
    gemm_bf16_kernel<<<dim3(1024 / BN, kT / BM), 256, 0, stream>>>(
        h_bf, wk_bf, k_f, nullptr, nullptr, nullptr, kT, 1024, kD, 0);
    gemm_bf16_kernel<<<dim3(kVD / BN, kT / BM), 256, 0, stream>>>(
        h_bf, wv_bf, nullptr, v_bf, nullptr, nullptr, kT, kVD, kD, 1);
    gemm_bf16_kernel<<<dim3(kVD / BN, kT / BM), 256, 0, stream>>>(
        h_bf, wg_bf, nullptr, g_bf, nullptr, nullptr, kT, kVD, kD, 1);

    // rotary (q gets DK^-0.5 fused)
    rotary_kernel<<<(kT * kH * 32) / 256, 256, 0, stream>>>(q_f, qr_bf, 0.125f);
    rotary_kernel<<<(kT * kH * 32) / 256, 256, 0, stream>>>(k_f, kr_bf, 1.0f);

    // retention scan
    retention_kernel<<<kH, 256, 0, stream>>>(qr_bf, kr_bf, v_bf, o_f);

    // groupnorm + silu gate
    gn_gate_kernel<<<(kT * kH) / 8, 256, 0, stream>>>(o_f, g_bf, gn_w + i * kDV, gated);

    // output projection
    gemm_bf16_kernel<<<dim3(kD / BN, kT / BM), 256, 0, stream>>>(
        gated, wo_bf, y_f, nullptr, nullptr, nullptr, kT, kD, kVD, 0);

    // LN2 -> h2 (f32 + bf16)
    layernorm_kernel<<<kT, 256, 0, stream>>>(y_f, ln2_w + i * kD, ln2_b + i * kD,
                                             h2_f, h2_bf);
    // FFN1: bias + exact GELU -> bf16
    gemm_bf16_kernel<<<dim3(kF / BN, kT / BM), 256, 0, stream>>>(
        h2_bf, w1_bf, nullptr, ffn1_bf, ffn_b1 + (size_t)i * kF, nullptr,
        kT, kF, kD, 2);
    // FFN2: bias + residual(h2) -> f32 (new x)
    gemm_bf16_kernel<<<dim3(kD / BN, kT / BM), 256, 0, stream>>>(
        ffn1_bf, w2_bf, xout, nullptr, ffn_b2 + (size_t)i * kD, h2_f,
        kT, kD, kF, 3);
  }
  (void)in_sizes; (void)n_in; (void)out_size; (void)ws_size;
}
